// TTTLinearAnalytic_11570641895783
// MI455X (gfx1250) — matrix operand order repaired and checked by reading, not compiled, not run on hardware
//
#include <hip/hip_runtime.h>
#include <hip/hip_bf16.h>
#include <math.h>

// ---------------------------------------------------------------------------
// TTT-linear forward for MI455X (gfx1250, wave32, WMMA)
//   B=4 S=2048 HID=1024 H=16 D=64 MB=16 NMB=128
// ---------------------------------------------------------------------------

typedef __bf16 bf16;
typedef __attribute__((ext_vector_type(16))) __bf16 v16bf;
typedef __attribute__((ext_vector_type(8)))  __bf16 v8bf;
typedef __attribute__((ext_vector_type(8)))  float  v8f;
typedef __attribute__((ext_vector_type(2)))  float  v2f;

union ABu { v16bf f; v8bf h[2]; };

// Async global->LDS copy via inline asm (CDNA5 GLOBAL_LOAD_ASYNC_TO_LDS_B128).
// LDS operand is the 32-bit LDS byte offset = low 32 bits of a generic LDS
// pointer (ISA 10.2: LDS aperture address truncates to addr[31:0]).
#define USE_ASYNC_ASM 1

__device__ __forceinline__ void async_copy_b128(const bf16* gsrc, bf16* ldst) {
#if USE_ASYNC_ASM
  asm volatile("global_load_async_to_lds_b128 %0, %1, off"
               :
               : "v"((unsigned)(uintptr_t)ldst), "v"(gsrc)
               : "memory");
#else
  *(v8bf*)ldst = *(const v8bf*)gsrc;
#endif
}

__device__ __forceinline__ void async_wait0() {
#if USE_ASYNC_ASM
  asm volatile("s_wait_asynccnt 0x0" ::: "memory");
#endif
}

__device__ __forceinline__ v8f zero8() {
  v8f z;
#pragma unroll
  for (int i = 0; i < 8; ++i) z[i] = 0.f;
  return z;
}

// ---------------------------------------------------------------------------
// bf16 WMMA GEMM core with LDS-staged B panel.
// One 256-thread block = 8 waves; each wave owns a 16x64 tile of C = A @ Bw^T.
// All 8 waves share the same 64 weight rows of Bw, so the B panel is staged
// through LDS (async DMA), 4 K-panels of 256.
//   A : M x 1024 row-major bf16,  Bw : N x 1024 row-major bf16 (weight)
// A-frag (16-bit 16x32): lane m=l&15; chunks at k0+8*hi and k0+16+8*hi.
// B-frag (16-bit 32x16): lane n=l&15; chunks at k0+8*hi and k0+16+8*hi.
// ---------------------------------------------------------------------------
#define BRS 264   // LDS row stride (bf16): 528 B -> bank step 4, 16B aligned

__device__ __forceinline__ void wave_gemm_16x64(const bf16* __restrict__ A,
                                                const bf16* __restrict__ Bw,
                                                int rowBase, int colStrip,
                                                v8f acc[4], bf16* Bsh) {
  const int tid  = threadIdx.x;
  const int lane = tid & 31;
  const int l15  = lane & 15;
  const int hi   = lane >> 4;
  const bf16* arow = A + (size_t)(rowBase + l15) * 1024;
#pragma unroll
  for (int t = 0; t < 4; ++t) acc[t] = zero8();

  for (int p = 0; p < 4; ++p) {
    // ---- stage 64 x 256 B-panel into LDS (each thread: 128 B = 8x b128) ----
    const bf16* grow = Bw + (size_t)(colStrip + (tid >> 2)) * 1024 + p * 256 + (tid & 3) * 64;
    bf16* lrow = Bsh + (tid >> 2) * BRS + (tid & 3) * 64;
#pragma unroll
    for (int i = 0; i < 64; i += 8) async_copy_b128(grow + i, lrow + i);
    async_wait0();
    __syncthreads();

    // ---- consume panel ----
    for (int kk = 0; kk < 256; kk += 32) {
      const int k0 = p * 256 + kk;
      __builtin_prefetch(arow + k0 + 288, 0, 0);   // global_prefetch_b8
      ABu a;
      a.h[0] = *(const v8bf*)(arow + k0 + 8 * hi);
      a.h[1] = *(const v8bf*)(arow + k0 + 16 + 8 * hi);
#pragma unroll
      for (int t = 0; t < 4; ++t) {
        const bf16* brow = Bsh + (16 * t + l15) * BRS + kk + 8 * hi;
        ABu bb;
        bb.h[0] = *(const v8bf*)brow;
        bb.h[1] = *(const v8bf*)(brow + 16);
        acc[t] = __builtin_amdgcn_wmma_f32_16x16x32_bf16(
            false, a.f, false, bb.f, (short)0, acc[t], false, false);
      }
    }
    __syncthreads();
  }
}

// ---------------------------------------------------------------------------
// fp32 -> bf16 conversion
// ---------------------------------------------------------------------------
__global__ void cvt_bf16_kernel(const float* __restrict__ in, bf16* __restrict__ out, int n) {
  int i = blockIdx.x * blockDim.x + threadIdx.x;
  if (i < n) out[i] = (bf16)in[i];
}

// ---------------------------------------------------------------------------
// eta = BASE_LR * clip(1/(m+1)+off, 0) * sigmoid(x . lr_w + lr_b) / D
// one block per (b,s); output layout [nmb][b][h][m]
// ---------------------------------------------------------------------------
__global__ __launch_bounds__(256) void lr_eta_kernel(const float* __restrict__ x,
                                                     const float* __restrict__ lr_w,
                                                     const float* __restrict__ lr_b,
                                                     const float* __restrict__ tok_off,
                                                     float* __restrict__ eta) {
  __shared__ float red[256];
  const int bs = blockIdx.x;
  const int t  = threadIdx.x;
  const int h  = t >> 4, seg = t & 15;
  const float* xr = x + (size_t)bs * 1024 + seg * 64;
  const float* wr = lr_w + (size_t)h * 1024 + seg * 64;
  float s = 0.f;
  for (int i = 0; i < 64; ++i) s += xr[i] * wr[i];
  red[t] = s;
  __syncthreads();
  if (t < 16) {
    float tot = 0.f;
    for (int g = 0; g < 16; ++g) tot += red[t * 16 + g];
    const float lv = 1.f / (1.f + expf(-(tot + lr_b[t])));
    const int b = bs >> 11, sIdx = bs & 2047;
    const int nmb = sIdx >> 4, m = sIdx & 15;
    float tok = 1.f / (float)(m + 1) + tok_off[m];
    tok = tok > 0.f ? tok : 0.f;
    eta[(((size_t)nmb * 4 + b) * 16 + t) * 16 + m] = tok * lv * (1.f / 64.f);
  }
}

// ---------------------------------------------------------------------------
// QKV GEMM, scattering into scan layout [nmb][b][h][m][d]
// ---------------------------------------------------------------------------
__global__ __launch_bounds__(256) void gemm_qkv_scan_kernel(const bf16* __restrict__ A,
                                                            const bf16* __restrict__ Bw,
                                                            float* __restrict__ Out) {
  __shared__ bf16 Bsh[64 * BRS];
  const int wave = threadIdx.x >> 5;
  const int lane = threadIdx.x & 31;
  const int l15 = lane & 15, hi = lane >> 4;
  const int rowBase  = blockIdx.x * 128 + wave * 16;
  const int colStrip = blockIdx.y * 64;
  v8f acc[4];
  wave_gemm_16x64(A, Bw, rowBase, colStrip, acc, Bsh);
#pragma unroll
  for (int t = 0; t < 4; ++t) {
    const int col = colStrip + 16 * t + l15;
    const int hh = col >> 6, d = col & 63;
#pragma unroll
    for (int r = 0; r < 8; ++r) {
      const int row = rowBase + 8 * hi + r;
      const int b = row >> 11, s = row & 2047;
      const int nmb = s >> 4, m = s & 15;
      Out[((((size_t)nmb * 4 + b) * 16 + hh) * 16 + m) * 64 + d] = acc[t][r];
    }
  }
}

// ---------------------------------------------------------------------------
// Post-scan per-row LN stats of y (HID=1024 per row)
// ---------------------------------------------------------------------------
__global__ __launch_bounds__(256) void row_stats_kernel(const float* __restrict__ y,
                                                        float* __restrict__ stats) {
  __shared__ float s1[256], s2[256];
  const int row = blockIdx.x, t = threadIdx.x;
  const float* p = y + (size_t)row * 1024;
  float a = 0.f, b = 0.f;
  for (int i = t; i < 1024; i += 256) { float v = p[i]; a += v; b += v * v; }
  s1[t] = a; s2[t] = b;
  __syncthreads();
  for (int off = 128; off > 0; off >>= 1) {
    if (t < off) { s1[t] += s1[t + off]; s2[t] += s2[t + off]; }
    __syncthreads();
  }
  if (t == 0) {
    const float mu = s1[0] * (1.f / 1024.f);
    const float var = s2[0] * (1.f / 1024.f) - mu * mu;
    stats[2 * row] = mu;
    stats[2 * row + 1] = rsqrtf(var + 1e-5f);
  }
}

// ---------------------------------------------------------------------------
// gate GEMM: t = gelu(x@Wg^T) * ((y-mu)*rstd*post_w + post_b)  -> bf16
// ---------------------------------------------------------------------------
__global__ __launch_bounds__(256) void gemm_gate_kernel(const bf16* __restrict__ A,
                                                        const bf16* __restrict__ Bw,
                                                        const float* __restrict__ y,
                                                        const float* __restrict__ stats,
                                                        const float* __restrict__ pw,
                                                        const float* __restrict__ pb,
                                                        bf16* __restrict__ Out) {
  __shared__ bf16 Bsh[64 * BRS];
  const int wave = threadIdx.x >> 5;
  const int lane = threadIdx.x & 31;
  const int l15 = lane & 15, hi = lane >> 4;
  const int rowBase  = blockIdx.x * 128 + wave * 16;
  const int colStrip = blockIdx.y * 64;
  v8f acc[4];
  wave_gemm_16x64(A, Bw, rowBase, colStrip, acc, Bsh);
#pragma unroll
  for (int t = 0; t < 4; ++t) {
    const int col = colStrip + 16 * t + l15;
    const float gw = pw[col], gb = pb[col];
#pragma unroll
    for (int r = 0; r < 8; ++r) {
      const int row = rowBase + 8 * hi + r;
      const float g  = acc[t][r];
      const float ge = 0.5f * g * (1.f + erff(g * 0.70710678118f));   // exact gelu
      const float mu = stats[2 * row], rs = stats[2 * row + 1];
      const float yl = (y[(size_t)row * 1024 + col] - mu) * rs * gw + gb;
      Out[(size_t)row * 1024 + col] = (bf16)(ge * yl);
    }
  }
}

// ---------------------------------------------------------------------------
// final GEMM: out = t @ Wo^T (fp32 out, plain layout)
// ---------------------------------------------------------------------------
__global__ __launch_bounds__(256) void gemm_out_kernel(const bf16* __restrict__ A,
                                                       const bf16* __restrict__ Bw,
                                                       float* __restrict__ Out) {
  __shared__ bf16 Bsh[64 * BRS];
  const int wave = threadIdx.x >> 5;
  const int lane = threadIdx.x & 31;
  const int l15 = lane & 15, hi = lane >> 4;
  const int rowBase  = blockIdx.x * 128 + wave * 16;
  const int colStrip = blockIdx.y * 64;
  v8f acc[4];
  wave_gemm_16x64(A, Bw, rowBase, colStrip, acc, Bsh);
#pragma unroll
  for (int t = 0; t < 4; ++t) {
    const int col = colStrip + 16 * t + l15;
#pragma unroll
    for (int r = 0; r < 8; ++r) {
      const int row = rowBase + 8 * hi + r;
      Out[(size_t)row * 1024 + col] = acc[t][r];
    }
  }
}

// ---------------------------------------------------------------------------
// TTT scan: one block per head (16 blocks, 128 threads = 4 waves = 4 batches).
// State W(64x64), b(64) live in LDS. Small matmuls via fp32 WMMA 16x16x4:
//   Z  = Km@W + b
//   gZ = ln_vjp(Z, ln(Z)-(V-K))
//   M  = tril(Qm@Km^T + 1)      (folds Attn and b_bar terms)
//   Zb = Qm@W + b - eta .* (M @ gZ)
//   out = Qm + ln(Zb)
//   W -= mean_b Km[15]^T (eta15*gZ[15]);  b -= mean_b eta15*gZ[15]
// ---------------------------------------------------------------------------
__global__ __launch_bounds__(128) void ttt_scan_kernel(const float* __restrict__ Qs,
                                                       const float* __restrict__ Ks,
                                                       const float* __restrict__ Vs,
                                                       const float* __restrict__ eta,
                                                       const float* __restrict__ W_init,
                                                       const float* __restrict__ b_init,
                                                       const float* __restrict__ ttt_w,
                                                       const float* __restrict__ ttt_b,
                                                       float* __restrict__ out) {
  const int h    = blockIdx.x;
  const int tid  = threadIdx.x;
  const int wave = tid >> 5;          // batch element
  const int lane = tid & 31;
  const int l15  = lane & 15;
  const int hi   = lane >> 4;

  __shared__ float Wsh[64][65];
  __shared__ float bsh[64];
  __shared__ float gsh[64];
  __shared__ float btsh[64];
  __shared__ float gZsh[4][16][65];
  __shared__ float Msh[4][16][17];
  __shared__ float kvsh[4][64];
  __shared__ float gvsh[4][64];
  __shared__ float etash[4][16];

  for (int i = tid; i < 64 * 64; i += 128) Wsh[i >> 6][i & 63] = W_init[h * 4096 + i];
  if (tid < 64) {
    bsh[tid]  = b_init[h * 64 + tid];
    gsh[tid]  = ttt_w[h * 64 + tid];
    btsh[tid] = ttt_b[h * 64 + tid];
  }
  __syncthreads();

  for (int step = 0; step < 128; ++step) {
    const size_t ibh = ((size_t)(step * 4 + wave) * 16 + h) * 16;
    const float* Kp = Ks + ibh * 64;
    const float* Qp = Qs + ibh * 64;
    const float* Vp = Vs + ibh * 64;
    if (lane < 16) etash[wave][lane] = eta[ibh + lane];

    // prefetch next step's K/Q/V tiles (4 KB each: one 128B line per lane)
    if (step + 1 < 128) {
      const size_t nb = ((size_t)((step + 1) * 4 + wave) * 16 + h) * 16 * 64;
      __builtin_prefetch(Ks + nb + lane * 32, 0, 0);
      __builtin_prefetch(Qs + nb + lane * 32, 0, 0);
      __builtin_prefetch(Vs + nb + lane * 32, 0, 0);
    }

    // A-fragment caches for Km / Qm (f32 16x4 layout: v0=K[4kc+2hi], v1=K[4kc+2hi+1])
    v2f ka[16], qa[16];
#pragma unroll
    for (int kc = 0; kc < 16; ++kc) {
      const float2 k2 = *(const float2*)(Kp + l15 * 64 + 4 * kc + 2 * hi);
      const float2 q2 = *(const float2*)(Qp + l15 * 64 + 4 * kc + 2 * hi);
      ka[kc][0] = k2.x; ka[kc][1] = k2.y;
      qa[kc][0] = q2.x; qa[kc][1] = q2.y;
    }

    // ---- Z = Km @ W + b ----
    v8f accZ[4];
#pragma unroll
    for (int t = 0; t < 4; ++t) accZ[t] = zero8();
    for (int kc = 0; kc < 16; ++kc) {
      const int k = 4 * kc + 2 * hi;
#pragma unroll
      for (int t = 0; t < 4; ++t) {
        v2f bv; const int c = 16 * t + l15;
        bv[0] = Wsh[k][c]; bv[1] = Wsh[k + 1][c];
        accZ[t] = __builtin_amdgcn_wmma_f32_16x16x4_f32(
            false, ka[kc], false, bv, (short)0, accZ[t], false, false);
      }
    }
#pragma unroll
    for (int t = 0; t < 4; ++t) {
      const float bb = bsh[16 * t + l15];
#pragma unroll
      for (int r = 0; r < 8; ++r) accZ[t][r] += bb;
    }

    // ---- LN stats over D=64 per row (row m = 8*hi + r) ----
    float mu[8], rstd[8];
#pragma unroll
    for (int r = 0; r < 8; ++r) {
      float s1 = 0.f, s2 = 0.f;
#pragma unroll
      for (int t = 0; t < 4; ++t) { const float z = accZ[t][r]; s1 += z; s2 += z * z; }
#pragma unroll
      for (int msk = 1; msk < 16; msk <<= 1) {
        s1 += __shfl_xor(s1, msk, 32);
        s2 += __shfl_xor(s2, msk, 32);
      }
      mu[r] = s1 * (1.f / 64.f);
      rstd[r] = rsqrtf(s2 * (1.f / 64.f) - mu[r] * mu[r] + 1e-5f);
    }

    // ---- gZ = ln_vjp(Z, ln(Z) - (V - K)) ----
    float gxn[4][8];
    float Sgx[8], Sgxc[8], Sxc[8];
#pragma unroll
    for (int r = 0; r < 8; ++r) { Sgx[r] = 0.f; Sgxc[r] = 0.f; Sxc[r] = 0.f; }
#pragma unroll
    for (int t = 0; t < 4; ++t) {
      const int c = 16 * t + l15;
      const float gc = gsh[c], bc = btsh[c];
#pragma unroll
      for (int r = 0; r < 8; ++r) {
        const int m = 8 * hi + r;
        const float xc = accZ[t][r] - mu[r];
        const float kk = Kp[m * 64 + c];
        const float vv = Vp[m * 64 + c];
        const float go = (xc * rstd[r]) * gc + bc - (vv - kk);
        const float gx = go * gc;
        accZ[t][r] = xc;      // reuse as centered value
        gxn[t][r]  = gx;
        Sgx[r] += gx; Sgxc[r] += gx * xc; Sxc[r] += xc;
      }
    }
#pragma unroll
    for (int r = 0; r < 8; ++r) {
#pragma unroll
      for (int msk = 1; msk < 16; msk <<= 1) {
        Sgx[r]  += __shfl_xor(Sgx[r],  msk, 32);
        Sgxc[r] += __shfl_xor(Sgxc[r], msk, 32);
        Sxc[r]  += __shfl_xor(Sxc[r],  msk, 32);
      }
      const float r3   = rstd[r] * rstd[r] * rstd[r];
      const float gvar = Sgxc[r] * -0.5f * r3;
      const float gmu  = Sgx[r] * (-rstd[r]) + gvar * (-2.f / 64.f) * Sxc[r];
      Sgx[r]  = gmu;
      Sgxc[r] = gvar;
    }
#pragma unroll
    for (int t = 0; t < 4; ++t) {
#pragma unroll
      for (int r = 0; r < 8; ++r) {
        const float gz = gxn[t][r] * rstd[r] + Sgxc[r] * (2.f / 64.f) * accZ[t][r]
                       + Sgx[r] * (1.f / 64.f);
        gZsh[wave][8 * hi + r][16 * t + l15] = gz;
      }
    }

    // ---- M = tril(Qm @ Km^T + 1) (B-frag of Km^T == A-frag of Km) ----
    v8f accA = zero8();
    for (int kc = 0; kc < 16; ++kc)
      accA = __builtin_amdgcn_wmma_f32_16x16x4_f32(
          false, qa[kc], false, ka[kc], (short)0, accA, false, false);
#pragma unroll
    for (int r = 0; r < 8; ++r) {
      const int m = 8 * hi + r;
      Msh[wave][m][l15] = (l15 <= m) ? (accA[r] + 1.0f) : 0.0f;
    }

    // ---- Zb = Qm@W + b - eta .* (M @ gZ) ----
    v8f accQ[4], accC[4];
#pragma unroll
    for (int t = 0; t < 4; ++t) { accQ[t] = zero8(); accC[t] = zero8(); }
    for (int kc = 0; kc < 16; ++kc) {
      const int k = 4 * kc + 2 * hi;
#pragma unroll
      for (int t = 0; t < 4; ++t) {
        v2f bv; const int c = 16 * t + l15;
        bv[0] = Wsh[k][c]; bv[1] = Wsh[k + 1][c];
        accQ[t] = __builtin_amdgcn_wmma_f32_16x16x4_f32(
            false, qa[kc], false, bv, (short)0, accQ[t], false, false);
      }
    }
#pragma unroll
    for (int kc = 0; kc < 4; ++kc) {
      const int k = 4 * kc + 2 * hi;
      v2f am; am[0] = Msh[wave][l15][k]; am[1] = Msh[wave][l15][k + 1];
#pragma unroll
      for (int t = 0; t < 4; ++t) {
        v2f bv; const int c = 16 * t + l15;
        bv[0] = gZsh[wave][k][c]; bv[1] = gZsh[wave][k + 1][c];
        accC[t] = __builtin_amdgcn_wmma_f32_16x16x4_f32(
            false, am, false, bv, (short)0, accC[t], false, false);
      }
    }
    float em[8];
#pragma unroll
    for (int r = 0; r < 8; ++r) em[r] = etash[wave][8 * hi + r];
#pragma unroll
    for (int t = 0; t < 4; ++t) {
      const float bb = bsh[16 * t + l15];
#pragma unroll
      for (int r = 0; r < 8; ++r)
        accQ[t][r] = accQ[t][r] + bb - em[r] * accC[t][r];
    }

    // ---- out = Qm + ln(Zb) ----
#pragma unroll
    for (int r = 0; r < 8; ++r) {
      float s1 = 0.f, s2 = 0.f;
#pragma unroll
      for (int t = 0; t < 4; ++t) { const float z = accQ[t][r]; s1 += z; s2 += z * z; }
#pragma unroll
      for (int msk = 1; msk < 16; msk <<= 1) {
        s1 += __shfl_xor(s1, msk, 32);
        s2 += __shfl_xor(s2, msk, 32);
      }
      const float mu2 = s1 * (1.f / 64.f);
      mu[r] = mu2;
      rstd[r] = rsqrtf(s2 * (1.f / 64.f) - mu2 * mu2 + 1e-5f);
    }
#pragma unroll
    for (int t = 0; t < 4; ++t) {
      const int c = 16 * t + l15;
      const float gc = gsh[c], bc = btsh[c];
#pragma unroll
      for (int r = 0; r < 8; ++r) {
        const int m = 8 * hi + r;
        const float o = Qp[m * 64 + c] + (accQ[t][r] - mu[r]) * rstd[r] * gc + bc;
        out[((size_t)wave * 2048 + step * 16 + m) * 1024 + h * 64 + c] = o;
      }
    }

    // ---- state update (rank-1 per batch, averaged across batches) ----
    {
      const float el = etash[wave][15];
      for (int e = lane; e < 64; e += 32) {
        kvsh[wave][e] = Kp[15 * 64 + e];
        gvsh[wave][e] = el * gZsh[wave][15][e];
      }
    }
    __syncthreads();
    for (int i = tid; i < 4096; i += 128) {
      const int d = i >> 6, e2 = i & 63;
      float acc = 0.f;
#pragma unroll
      for (int b = 0; b < 4; ++b) acc += kvsh[b][d] * gvsh[b][e2];
      Wsh[d][e2] -= 0.25f * acc;
    }
    if (tid < 64) {
      float acc = 0.f;
#pragma unroll
      for (int b = 0; b < 4; ++b) acc += gvsh[b][tid];
      bsh[tid] -= 0.25f * acc;
    }
    __syncthreads();
  }
}

// ---------------------------------------------------------------------------
extern "C" void kernel_launch(void* const* d_in, const int* in_sizes, int n_in,
                              void* d_out, int out_size, void* d_ws, size_t ws_size,
                              hipStream_t stream) {
  (void)in_sizes; (void)n_in; (void)out_size; (void)ws_size;
  const float* x    = (const float*)d_in[0];
  const float* Wq   = (const float*)d_in[1];
  const float* Wk   = (const float*)d_in[2];
  const float* Wv   = (const float*)d_in[3];
  const float* Wi   = (const float*)d_in[4];
  const float* bi   = (const float*)d_in[5];
  const float* tw   = (const float*)d_in[6];
  const float* tb   = (const float*)d_in[7];
  const float* pw   = (const float*)d_in[8];
  const float* pb   = (const float*)d_in[9];
  const float* lw   = (const float*)d_in[10];
  const float* lb   = (const float*)d_in[11];
  const float* toff = (const float*)d_in[12];
  const float* Wg   = (const float*)d_in[13];
  const float* Wo   = (const float*)d_in[14];
  float* out = (float*)d_out;

  char* ws = (char*)d_ws;
  size_t off = 0;
  auto take = [&](size_t bytes) -> char* {
    char* p = ws + off;
    off = (off + bytes + 255) & ~(size_t)255;
    return p;
  };
  bf16*  xb   = (bf16*) take((size_t)8192 * 1024 * 2);
  bf16*  wqb  = (bf16*) take((size_t)1024 * 1024 * 2);
  bf16*  wkb  = (bf16*) take((size_t)1024 * 1024 * 2);
  bf16*  wvb  = (bf16*) take((size_t)1024 * 1024 * 2);
  bf16*  wgb  = (bf16*) take((size_t)1024 * 1024 * 2);
  bf16*  wob  = (bf16*) take((size_t)1024 * 1024 * 2);
  float* Qsc  = (float*)take((size_t)8192 * 1024 * 4);
  float* Ksc  = (float*)take((size_t)8192 * 1024 * 4);
  float* Vsc  = (float*)take((size_t)8192 * 1024 * 4);
  float* etaw = (float*)take((size_t)131072 * 4);
  float* st   = (float*)take((size_t)8192 * 2 * 4);
  bf16*  tbuf = (bf16*) take((size_t)8192 * 1024 * 2);

  cvt_bf16_kernel<<<(8388608 + 255) / 256, 256, 0, stream>>>(x,  xb,  8388608);
  cvt_bf16_kernel<<<4096, 256, 0, stream>>>(Wq, wqb, 1048576);
  cvt_bf16_kernel<<<4096, 256, 0, stream>>>(Wk, wkb, 1048576);
  cvt_bf16_kernel<<<4096, 256, 0, stream>>>(Wv, wvb, 1048576);
  cvt_bf16_kernel<<<4096, 256, 0, stream>>>(Wg, wgb, 1048576);
  cvt_bf16_kernel<<<4096, 256, 0, stream>>>(Wo, wob, 1048576);

  lr_eta_kernel<<<8192, 256, 0, stream>>>(x, lw, lb, toff, etaw);

  dim3 gg(64, 16);
  gemm_qkv_scan_kernel<<<gg, 256, 0, stream>>>(xb, wqb, Qsc);
  gemm_qkv_scan_kernel<<<gg, 256, 0, stream>>>(xb, wkb, Ksc);
  gemm_qkv_scan_kernel<<<gg, 256, 0, stream>>>(xb, wvb, Vsc);

  ttt_scan_kernel<<<16, 128, 0, stream>>>(Qsc, Ksc, Vsc, etaw, Wi, bi, tw, tb, out);

  row_stats_kernel<<<8192, 256, 0, stream>>>(out, st);
  gemm_gate_kernel<<<gg, 256, 0, stream>>>(xb, wgb, out, st, pw, pb, tbuf);
  gemm_out_kernel<<<gg, 256, 0, stream>>>(tbuf, wob, out);
}
